// MultilayerGRU_62947040690601
// MI455X (gfx1250) — compile-verified, matrix operand order checked by reading, not run on hardware
//
#include <hip/hip_runtime.h>
#include <hip/hip_bf16.h>
#include <cstdint>
#include <cstddef>

// Problem constants (match reference)
#define B_ 64
#define S_ 1024
#define I_ 128
#define H_ 512
#define O_ 128
#define L_ 2

#define SCAN_WGS 8          // persistent workgroups in the scan
#define SCAN_THREADS 256    // 8 waves (wave32)

typedef __attribute__((ext_vector_type(16))) __bf16 v16bf;
typedef __attribute__((ext_vector_type(8)))  __bf16 v8bf;
typedef __attribute__((ext_vector_type(8)))  float  v8f;

// ---------------------------------------------------------------------------
// CDNA5 async global->LDS copy (ASYNCcnt-tracked), GV addressing.
// ---------------------------------------------------------------------------
static __device__ __forceinline__ void async_copy_b128(unsigned lds_off,
                                                       const void* gaddr) {
  asm volatile("global_load_async_to_lds_b128 %0, %1, off"
               :: "v"(lds_off), "v"(gaddr)
               : "memory");
}
template <int N>
static __device__ __forceinline__ void wait_async_le() {
  asm volatile("s_wait_asynccnt %0" :: "i"(N) : "memory");
}

// ---------------------------------------------------------------------------
// WMMA helpers (CDNA5: D = A(16x32 bf16) * B(32x16 bf16) + C(16x16 f32))
// ---------------------------------------------------------------------------
static __device__ __forceinline__ v8f wmma_bf16(v16bf a, v16bf b, v8f c) {
  return __builtin_amdgcn_wmma_f32_16x16x32_bf16(false, a, false, b, (short)0, c,
                                                 false, false);
}

// 16x32 bf16 fragment (A layout; B identical on pre-transposed N x K weights).
static __device__ __forceinline__ v16bf load_frag(const __bf16* base, int ld, int lane) {
  const int m    = lane & 15;
  const int half = (lane >> 4) << 3;       // 0 or 8
  const __bf16* p = base + (size_t)m * (size_t)ld;
  v8bf lo = *(const v8bf*)(p + half);
  v8bf hi = *(const v8bf*)(p + 16 + half);
  v16bf r;
#pragma unroll
  for (int i = 0; i < 8; ++i) { r[i] = lo[i]; r[i + 8] = hi[i]; }
  return r;
}

// Fast activations: single v_exp_f32 + v_rcp_f32 (no IEEE div sequence).
static __device__ __forceinline__ float sigm(float x) {
  return __builtin_amdgcn_rcpf(1.0f + __expf(-x));
}
static __device__ __forceinline__ float tanh_fast(float x) {
  return 1.0f - 2.0f * __builtin_amdgcn_rcpf(1.0f + __expf(2.0f * x));
}

// ---------------------------------------------------------------------------
// Conversion / transpose prep kernels
// ---------------------------------------------------------------------------
__global__ void k_cvt_bf16(const float* __restrict__ src, __bf16* __restrict__ dst,
                           size_t n) {
  size_t i = (size_t)blockIdx.x * blockDim.x + threadIdx.x;
  if (i < n) dst[i] = (__bf16)src[i];
}

// W: K x N (row major, fp32)  ->  WT: N x K (row major, bf16)
__global__ void k_transpose_w(const float* __restrict__ W, __bf16* __restrict__ WT,
                              int K, int N) {
  size_t i = (size_t)blockIdx.x * blockDim.x + threadIdx.x;
  size_t total = (size_t)K * (size_t)N;
  if (i < total) {
    int k = (int)(i / (size_t)N);
    int n = (int)(i % (size_t)N);
    WT[(size_t)n * (size_t)K + (size_t)k] = (__bf16)W[i];
  }
}

// Init per-layer scan state: hcur (bf16) from hidden_state fp32, reset barrier.
__global__ void k_init_layer(const float* __restrict__ hidden, __bf16* __restrict__ hcur,
                             int layer, unsigned* __restrict__ bar) {
  int i = blockIdx.x * blockDim.x + threadIdx.x;
  if (i < B_ * H_) {
    int b = i >> 9;          // /H_
    int c = i & (H_ - 1);
    hcur[i] = (__bf16)hidden[((size_t)b * L_ + layer) * H_ + c];
  }
  if (i < 2) bar[i] = 0u;
}

// ---------------------------------------------------------------------------
// Tiled bf16 WMMA GEMM:  out[...] = A[M x K] * WT[N x K]^T + bias
// TIME_MAJOR: A rows are m = b*S + t but output row index is (t*B + b)
// (time-major pre-activations -> contiguous 64KB panel per scan step).
// ---------------------------------------------------------------------------
template <bool OUT_BF16, bool TIME_MAJOR>
static __device__ __forceinline__ void store_tile(void* out, const float* bias,
                                                  int m0, int n0, v8f acc, int N,
                                                  int lane) {
  int n    = n0 + (lane & 15);
  int moff = (lane >> 4) << 3;
  float bv = bias ? bias[n] : 0.0f;
#pragma unroll
  for (int v = 0; v < 8; ++v) {
    int m = m0 + moff + v;
    size_t row = TIME_MAJOR ? (size_t)((m & (S_ - 1)) * B_ + (m >> 10))
                            : (size_t)m;
    size_t idx = row * (size_t)N + (size_t)n;
    if (OUT_BF16) ((__bf16*)out)[idx] = (__bf16)(acc[v] + bv);
    else          ((float*)out)[idx]  = acc[v] + bv;
  }
}

template <bool OUT_BF16, bool TIME_MAJOR>
__global__ void __launch_bounds__(256)
k_gemm(const __bf16* __restrict__ A, const __bf16* __restrict__ WT,
       const float* __restrict__ bias, void* __restrict__ out,
       int M, int N, int K) {
  __shared__ __bf16 sA[64 * 32];
  __shared__ __bf16 sB[64 * 32];
  const int m0 = blockIdx.x * 64;
  const int n0 = blockIdx.y * 64;
  const int tid = threadIdx.x, lane = tid & 31, wv = tid >> 5;

  const int t0 = wv, t1 = wv + 8;                 // 2 of 16 tiles (4x4 grid)
  const int tm0 = (t0 >> 2) << 4, tn0 = (t0 & 3) << 4;
  const int tm1 = (t1 >> 2) << 4, tn1 = (t1 & 3) << 4;

  const int row = tid >> 2;
  const int seg = (tid & 3) << 3;  // 8 bf16 = 16B
  const unsigned dstA = (unsigned)(uintptr_t)&sA[row * 32 + seg];
  const unsigned dstB = (unsigned)(uintptr_t)&sB[row * 32 + seg];

  v8f acc0, acc1;
#pragma unroll
  for (int j = 0; j < 8; ++j) { acc0[j] = 0.0f; acc1[j] = 0.0f; }

  for (int k0 = 0; k0 < K; k0 += 32) {
    __syncthreads();
    async_copy_b128(dstA, &A[(size_t)(m0 + row) * (size_t)K + (size_t)(k0 + seg)]);
    async_copy_b128(dstB, &WT[(size_t)(n0 + row) * (size_t)K + (size_t)(k0 + seg)]);
    wait_async_le<0>();
    __syncthreads();
    v16bf b0 = load_frag(&sB[tn0 * 32], 32, lane);
    v16bf a0 = load_frag(&sA[tm0 * 32], 32, lane);
    acc0 = wmma_bf16(a0, b0, acc0);
    v16bf b1 = load_frag(&sB[tn1 * 32], 32, lane);
    v16bf a1 = load_frag(&sA[tm1 * 32], 32, lane);
    acc1 = wmma_bf16(a1, b1, acc1);
  }
  store_tile<OUT_BF16, TIME_MAJOR>(out, bias, m0 + tm0, n0 + tn0, acc0, N, lane);
  store_tile<OUT_BF16, TIME_MAJOR>(out, bias, m0 + tm1, n0 + tn1, acc1, N, lane);
}

// ---------------------------------------------------------------------------
// Device-scope sense-reversing barrier across the SCAN_WGS persistent WGs.
// ---------------------------------------------------------------------------
static __device__ __forceinline__ void global_barrier(unsigned* bar) {
  __syncthreads();
  if (threadIdx.x == 0) {
    __threadfence();
    unsigned gen = __hip_atomic_load(bar + 1, __ATOMIC_ACQUIRE, __HIP_MEMORY_SCOPE_AGENT);
    unsigned arr = __hip_atomic_fetch_add(bar, 1u, __ATOMIC_ACQ_REL, __HIP_MEMORY_SCOPE_AGENT);
    if (arr == (unsigned)(SCAN_WGS - 1)) {
      __hip_atomic_store(bar, 0u, __ATOMIC_RELAXED, __HIP_MEMORY_SCOPE_AGENT);
      __hip_atomic_fetch_add(bar + 1, 1u, __ATOMIC_ACQ_REL, __HIP_MEMORY_SCOPE_AGENT);
    } else {
      while (__hip_atomic_load(bar + 1, __ATOMIC_ACQUIRE, __HIP_MEMORY_SCOPE_AGENT) == gen) {
        __builtin_amdgcn_s_sleep(8);
      }
    }
  }
  __syncthreads();
}

// ---------------------------------------------------------------------------
// Persistent recurrent scan.  8 WGs; WG w owns hidden columns [w*64, w*64+64)
// of each gate; recurrent weight slices LDS-resident.
// Dynamic LDS: 192KB weights + 64KB stage + 16KB z + 16KB h + 24KB pa = 312KB
// ---------------------------------------------------------------------------
#define SCAN_LDS_BYTES (192*512*2 + 64*512*2 + 64*64*4 + 64*64*4 + 3*64*64*2)

// issue async copies for a full 64x512 bf16 activation panel (16 per thread)
static __device__ __forceinline__ void stage_panel_issue(__bf16* dstLDS,
                                                         const __bf16* src, int tid) {
  const unsigned base = (unsigned)(uintptr_t)dstLDS;
  const char* s = (const char*)src;
#pragma unroll
  for (int j = 0; j < (B_ * H_ * 2) / (16 * SCAN_THREADS); ++j) {  // 16 iters
    int i = tid + j * SCAN_THREADS;
    async_copy_b128(base + (unsigned)i * 16u, s + (size_t)i * 16u);
  }
}

__global__ void __launch_bounds__(SCAN_THREADS, 1)
k_scan(const __bf16* __restrict__ pa,    // [3][S][B][H] bf16 (time-major, bias folded)
       const __bf16* __restrict__ whT,   // [3][H][H] bf16 (N x K transposed)
       const float*  __restrict__ h0,    // hidden_state (B, L, H) fp32
       __bf16* __restrict__ hcur,        // [B][H] bf16 running hidden
       __bf16* __restrict__ sbuf,        // [B][H] bf16 r*h exchange
       __bf16* __restrict__ ybf,         // [B][S][H] bf16 layer output
       float*  __restrict__ outH,        // d_out hidden section (B, L, H)
       int layer,
       unsigned* __restrict__ bar) {
  extern __shared__ char smem[];
  __bf16* wt    = (__bf16*)smem;                                  // [192][512]
  __bf16* stage = (__bf16*)(smem + 192 * 512 * 2);                // [64][512]
  float*  z_own = (float*)(smem + 192 * 512 * 2 + 64 * 512 * 2);  // [64][64]
  float*  h_own = (float*)((char*)z_own + 64 * 64 * 4);           // [64][64]
  __bf16* pa_st = (__bf16*)((char*)h_own + 64 * 64 * 4);          // [3][64][64]

  const int tid  = threadIdx.x;
  const int lane = tid & 31;
  const int wv   = tid >> 5;        // 0..7
  const int wg   = blockIdx.x;      // 0..7
  const int colbase = wg * 64;

  // --- one-time: async-pull our weight slice into LDS (192 rows x 512) ------
  {
    const unsigned wbase = (unsigned)(uintptr_t)wt;
    for (int c = tid; c < 192 * 64; c += SCAN_THREADS) {   // 16B chunks
      int rr = c >> 6;            // 0..191  (= g*64 + nl)
      int cc = c & 63;
      int g  = rr >> 6;
      int nl = rr & 63;
      const char* src = (const char*)(whT + ((size_t)g * H_ +
                                             (size_t)(colbase + nl)) * H_) + cc * 16;
      async_copy_b128(wbase + (unsigned)c * 16u, src);
    }
    wait_async_le<0>();
  }
  for (int i = tid; i < 64 * 64; i += SCAN_THREADS) {
    int b = i >> 6, cl = i & 63;
    h_own[i] = h0[((size_t)b * L_ + layer) * H_ + (size_t)(colbase + cl)];
  }
  __syncthreads();

  for (int t = 0; t < S_; ++t) {
    // ---- issue async stages: h panel (16/thread) then pa slices (6/thread) --
    stage_panel_issue(stage, hcur, tid);
    {
      const unsigned pbase = (unsigned)(uintptr_t)pa_st;
#pragma unroll
      for (int j = 0; j < 6; ++j) {          // 3 gates * 64 rows * 8 chunks
        int c   = tid + j * SCAN_THREADS;    // 0..1535
        int g   = c >> 9;
        int rem = c & 511;
        int r   = rem >> 3;                  // batch row
        int kk  = rem & 7;                   // 16B chunk within 128B row slice
        const char* src = (const char*)(pa + (((size_t)g * S_ + t) * B_ + r) * H_ +
                                        colbase) + kk * 16;
        async_copy_b128(pbase + (unsigned)c * 16u, src);
      }
    }
    // ---- warm L2 for next step's panels (contiguous in time-major) ---------
    if (tid < 192) {
      int g = tid >> 6, r = tid & 63;
      __builtin_prefetch(pa + (((size_t)g * S_ + (t + 1)) * B_ + r) * H_ + colbase,
                         0, 1);
    }
    wait_async_le<6>();     // own h-panel chunks complete (in-order ASYNCcnt)
    __syncthreads();        // => everyone's h-panel complete; pa still streaming

    // ---- phase 1 GEMM: waves 0-3 -> z, waves 4-7 -> r ----------------------
    const int gate = wv >> 2;          // 0 = z, 1 = r
    const int ncl  = (wv & 3) << 4;    // local col tile base
    v8f acc[4];
#pragma unroll
    for (int m = 0; m < 4; ++m)
#pragma unroll
      for (int j = 0; j < 8; ++j) acc[m][j] = 0.0f;

    for (int k0 = 0; k0 < H_; k0 += 32) {
      v16bf bf = load_frag(&wt[(size_t)(gate * 64 + ncl) * H_ + k0], H_, lane);
#pragma unroll
      for (int mt = 0; mt < 4; ++mt) {
        v16bf af = load_frag(&stage[(size_t)(mt * 16) * H_ + k0], H_, lane);
        acc[mt] = wmma_bf16(af, bf, acc[mt]);
      }
    }
    wait_async_le<0>();     // own pa chunks complete
    __syncthreads();        // everyone's pa chunks complete
    {
      const int nloc = ncl + (lane & 15);
      const int ncolg = colbase + nloc;
      const int moff = (lane >> 4) << 3;
      // batched LDS gathers (same base reg + immediate offsets -> ds clause)
      float pv[4][8], hv[4][8];
#pragma unroll
      for (int mt = 0; mt < 4; ++mt)
#pragma unroll
        for (int v = 0; v < 8; ++v) {
          int b = mt * 16 + moff + v;
          pv[mt][v] = (float)pa_st[((gate << 6) + b) * 64 + nloc];
          hv[mt][v] = h_own[b * 64 + nloc];
        }
#pragma unroll
      for (int mt = 0; mt < 4; ++mt)
#pragma unroll
        for (int v = 0; v < 8; ++v) {
          int b = mt * 16 + moff + v;
          float g = sigm(pv[mt][v] + acc[mt][v]);
          if (gate == 0) {
            z_own[b * 64 + nloc] = g;                 // keep z in LDS
          } else {
            float sv = g * hv[mt][v];                 // s = r .* h
            sbuf[(size_t)b * H_ + ncolg] = (__bf16)sv;
          }
        }
    }
    global_barrier(bar);   // all WGs published their s slice

    // ---- stage s (full 64x512 bf16) ----------------------------------------
    stage_panel_issue(stage, sbuf, tid);
    wait_async_le<0>();
    __syncthreads();

    // ---- phase 2: g gate, 64x64 output over 8 waves ------------------------
    const int ncl2  = (wv & 3) << 4;
    const int mbase = (wv >> 2) << 5;   // 0 or 32
    v8f accg[2];
#pragma unroll
    for (int m = 0; m < 2; ++m)
#pragma unroll
      for (int j = 0; j < 8; ++j) accg[m][j] = 0.0f;

    for (int k0 = 0; k0 < H_; k0 += 32) {
      v16bf bf = load_frag(&wt[(size_t)(128 + ncl2) * H_ + k0], H_, lane);
#pragma unroll
      for (int mt = 0; mt < 2; ++mt) {
        v16bf af = load_frag(&stage[(size_t)(mbase + mt * 16) * H_ + k0], H_, lane);
        accg[mt] = wmma_bf16(af, bf, accg[mt]);
      }
    }
    {
      const int nloc = ncl2 + (lane & 15);
      const int ncolg = colbase + nloc;
      const int moff = (lane >> 4) << 3;
      float pv[2][8], hv[2][8], zv[2][8];
#pragma unroll
      for (int mt = 0; mt < 2; ++mt)
#pragma unroll
        for (int v = 0; v < 8; ++v) {
          int b = mbase + mt * 16 + moff + v;
          pv[mt][v] = (float)pa_st[((2 << 6) + b) * 64 + nloc];
          hv[mt][v] = h_own[b * 64 + nloc];
          zv[mt][v] = z_own[b * 64 + nloc];
        }
#pragma unroll
      for (int mt = 0; mt < 2; ++mt)
#pragma unroll
        for (int v = 0; v < 8; ++v) {
          int b = mbase + mt * 16 + moff + v;
          float gv = tanh_fast(pv[mt][v] + accg[mt][v]);
          float hn = zv[mt][v] * hv[mt][v] + (1.0f - zv[mt][v]) * gv;
          h_own[b * 64 + nloc] = hn;
          __bf16 hb = (__bf16)hn;
          hcur[(size_t)b * H_ + ncolg] = hb;
          ybf[((size_t)b * S_ + t) * H_ + ncolg] = hb;
        }
    }
    global_barrier(bar);   // h_t published for next step
  }

  // ---- final hidden state (fp32) ------------------------------------------
  {
    const int nloc  = ((wv & 3) << 4) + (lane & 15);
    const int mbase = (wv >> 2) << 5;
    const int moff  = (lane >> 4) << 3;
#pragma unroll
    for (int mt = 0; mt < 2; ++mt)
#pragma unroll
      for (int v = 0; v < 8; ++v) {
        int b = mbase + mt * 16 + moff + v;
        outH[((size_t)b * L_ + layer) * H_ + (size_t)(colbase + nloc)] =
            h_own[b * 64 + nloc];
      }
  }
}

// ---------------------------------------------------------------------------
// Host launcher
// ---------------------------------------------------------------------------
extern "C" void kernel_launch(void* const* d_in, const int* in_sizes, int n_in,
                              void* d_out, int out_size, void* d_ws, size_t ws_size,
                              hipStream_t stream) {
  (void)in_sizes; (void)n_in; (void)out_size; (void)ws_size;

  const float* x      = (const float*)d_in[0];
  const float* hstate = (const float*)d_in[1];
  const float* P[20];
  for (int i = 0; i < 20; ++i) P[i] = (const float*)d_in[2 + i];
  // P: 0:Wzx0 1:Wzh0 2:bz0 3:Wrx0 4:Wrh0 5:br0 6:Wgx0 7:Wgh0 8:bg0
  //    9:Wzx1 10:Wzh1 11:bz1 12:Wrx1 13:Wrh1 14:br1 15:Wgx1 16:Wgh1 17:bg1
  //    18:Wout 19:bout

  char* ws = (char*)d_ws;
  size_t off = 0;
  auto carve = [&](size_t bytes) -> char* {
    char* p = ws + off;
    off += (bytes + 255) & ~(size_t)255;
    return p;
  };
  __bf16* x0bf = (__bf16*)carve((size_t)B_ * S_ * I_ * 2);
  __bf16* y0bf = (__bf16*)carve((size_t)B_ * S_ * H_ * 2);
  __bf16* y1bf = (__bf16*)carve((size_t)B_ * S_ * H_ * 2);
  __bf16* pa   = (__bf16*)carve((size_t)3 * B_ * S_ * H_ * 2);  // [3][S][B][H]
  __bf16* wxT0 = (__bf16*)carve((size_t)3 * H_ * I_ * 2);
  __bf16* whT0 = (__bf16*)carve((size_t)3 * H_ * H_ * 2);
  __bf16* wxT1 = (__bf16*)carve((size_t)3 * H_ * H_ * 2);
  __bf16* whT1 = (__bf16*)carve((size_t)3 * H_ * H_ * 2);
  __bf16* woT  = (__bf16*)carve((size_t)O_ * H_ * 2);
  __bf16* hcur = (__bf16*)carve((size_t)B_ * H_ * 2);
  __bf16* sbuf = (__bf16*)carve((size_t)B_ * H_ * 2);
  unsigned* bar = (unsigned*)carve(256);

  float* outY = (float*)d_out;                      // (B,S,O)
  float* outH = outY + (size_t)B_ * S_ * O_;        // (B,L,H)

  hipFuncSetAttribute((const void*)k_scan,
                      hipFuncAttributeMaxDynamicSharedMemorySize, SCAN_LDS_BYTES);

  const int M = B_ * S_;  // 65536

  // ---- convert activations ----
  {
    size_t n = (size_t)B_ * S_ * I_;
    k_cvt_bf16<<<dim3((unsigned)((n + 255) / 256)), dim3(256), 0, stream>>>(x, x0bf, n);
  }
  // ---- transpose + convert all weights ----
  auto launch_tr = [&](const float* W, __bf16* WT, int K, int N) {
    size_t tot = (size_t)K * N;
    k_transpose_w<<<dim3((unsigned)((tot + 255) / 256)), dim3(256), 0, stream>>>(W, WT, K, N);
  };
  launch_tr(P[0], wxT0 + 0 * (size_t)H_ * I_, I_, H_);   // Wzx0
  launch_tr(P[3], wxT0 + 1 * (size_t)H_ * I_, I_, H_);   // Wrx0
  launch_tr(P[6], wxT0 + 2 * (size_t)H_ * I_, I_, H_);   // Wgx0
  launch_tr(P[1], whT0 + 0 * (size_t)H_ * H_, H_, H_);   // Wzh0
  launch_tr(P[4], whT0 + 1 * (size_t)H_ * H_, H_, H_);   // Wrh0
  launch_tr(P[7], whT0 + 2 * (size_t)H_ * H_, H_, H_);   // Wgh0
  launch_tr(P[9],  wxT1 + 0 * (size_t)H_ * H_, H_, H_);  // Wzx1
  launch_tr(P[12], wxT1 + 1 * (size_t)H_ * H_, H_, H_);  // Wrx1
  launch_tr(P[15], wxT1 + 2 * (size_t)H_ * H_, H_, H_);  // Wgx1
  launch_tr(P[10], whT1 + 0 * (size_t)H_ * H_, H_, H_);  // Wzh1
  launch_tr(P[13], whT1 + 1 * (size_t)H_ * H_, H_, H_);  // Wrh1
  launch_tr(P[16], whT1 + 2 * (size_t)H_ * H_, H_, H_);  // Wgh1
  launch_tr(P[18], woT, H_, O_);                         // Wout

  // ---- layer 0: input projections (bias folded, time-major output) ----
  {
    dim3 grid(M / 64, H_ / 64);
    const float* bs[3] = {P[2], P[5], P[8]};
    for (int g = 0; g < 3; ++g)
      k_gemm<true, true><<<grid, dim3(256), 0, stream>>>(
          x0bf, wxT0 + (size_t)g * H_ * I_, bs[g],
          (void*)(pa + (size_t)g * B_ * S_ * H_), M, H_, I_);
  }
  // ---- layer 0: scan ----
  k_init_layer<<<dim3((B_ * H_ + 255) / 256), dim3(256), 0, stream>>>(hstate, hcur, 0, bar);
  k_scan<<<dim3(SCAN_WGS), dim3(SCAN_THREADS), SCAN_LDS_BYTES, stream>>>(
      pa, whT0, hstate, hcur, sbuf, y0bf, outH, 0, bar);

  // ---- layer 1: input projections ----
  {
    dim3 grid(M / 64, H_ / 64);
    const float* bs[3] = {P[11], P[14], P[17]};
    for (int g = 0; g < 3; ++g)
      k_gemm<true, true><<<grid, dim3(256), 0, stream>>>(
          y0bf, wxT1 + (size_t)g * H_ * H_, bs[g],
          (void*)(pa + (size_t)g * B_ * S_ * H_), M, H_, H_);
  }
  // ---- layer 1: scan ----
  k_init_layer<<<dim3((B_ * H_ + 255) / 256), dim3(256), 0, stream>>>(hstate, hcur, 1, bar);
  k_scan<<<dim3(SCAN_WGS), dim3(SCAN_THREADS), SCAN_LDS_BYTES, stream>>>(
      pa, whT1, hstate, hcur, sbuf, y1bf, outH, 1, bar);

  // ---- output projection (fp32 out) ----
  {
    dim3 grid(M / 64, O_ / 64);
    k_gemm<false, false><<<grid, dim3(256), 0, stream>>>(y1bf, woT, P[19], (void*)outY,
                                                         M, O_, H_);
  }
}
